// Up_21010980012359
// MI455X (gfx1250) — matrix operand order repaired and checked by reading, not compiled, not run on hardware
//
#include <hip/hip_runtime.h>

// ---------------------------------------------------------------------------
// Types
// ---------------------------------------------------------------------------
typedef __attribute__((ext_vector_type(16))) __bf16        v16bf;
typedef __attribute__((ext_vector_type(8)))  float         v8f;
typedef __attribute__((ext_vector_type(4)))  unsigned int  u32x4;
typedef __attribute__((ext_vector_type(4)))  float         f32x4;
typedef unsigned int u32;
typedef unsigned short u16;

// Problem constants
#define BB   4
#define CIN  128
#define HH   128
#define WW   128
#define CH   64
#define COUT 64
#define H2   256
#define W2   256
#define HP   258   // padded (halo 1)
#define WP   258

// LDS row strides (ushorts), padded to break ds_load_b128 bank conflicts:
// 72 = 36 dwords (gcd(36,64)=4 -> 16-lane bank rotation), 136 = 68 dwords.
#define RS64  72    // rows holding 64 bf16 (conv activations / conv weights)
#define RS128 136   // rows holding 128 bf16 (up weights / x1 tile)

static __device__ __forceinline__ u16 f32_to_bf16(float f) {
    u32 u = __float_as_uint(f);
    u32 r = u + 0x7FFFu + ((u >> 16) & 1u);   // round-to-nearest-even
    return (u16)(r >> 16);
}

union FragBF { v16bf v; u32x4 q[2]; };

// CDNA5 async copy: 16B global -> LDS, tracked by ASYNCcnt (no VGPR landing).
static __device__ __forceinline__ void async_copy16(u32 lds_byte_off, const void* g) {
    asm volatile("global_load_async_to_lds_b128 %0, %1, off"
                 :: "v"(lds_byte_off), "v"((unsigned long long)(size_t)g)
                 : "memory");
}
static __device__ __forceinline__ void wait_async0() {
    asm volatile("s_wait_asynccnt 0" ::: "memory");
}
static __device__ __forceinline__ u32 lds_off(const void* p) {
    return (u32)(size_t)p;   // LDS aperture: addr[31:0] is the LDS byte offset
}

// ---------------------------------------------------------------------------
// Zero the 1-pixel halo of both padded NHWC bf16 buffers (SAME padding).
// ---------------------------------------------------------------------------
__global__ __launch_bounds__(256)
void zero_halo(u16* __restrict__ m0, u16* __restrict__ m1) {
    u32 idx = blockIdx.x * 256u + threadIdx.x;
    u32 d  = idx & 31u;
    u32 pk = idx >> 5;
    if (pk >= 4u * 1028u) return;
    u32 b = pk / 1028u, k = pk % 1028u;
    u32 r, c;
    if      (k < 258u) { r = 0;   c = k; }
    else if (k < 516u) { r = 257; c = k - 258u; }
    else if (k < 774u) { r = k - 516u + 1u; c = 0; }
    else               { r = k - 774u + 1u; c = 257; }
    size_t off = (((size_t)b * HP + r) * WP + c) * 32 + d;  // dword index
    ((u32*)m0)[off] = 0u;
    ((u32*)m1)[off] = 0u;
}

// ---------------------------------------------------------------------------
// Repack weights to bf16 with padded row strides:
//   w1t/w2t: [tap(9)][oc(64)][RS64]    from torch [oc][ic][ky][kx]
//   wupT   : [pq(4)][co(64)][RS128]    from torch [ci][co][p][q]
// ---------------------------------------------------------------------------
__global__ __launch_bounds__(256)
void prep_weights(const float* __restrict__ w_up,
                  const float* __restrict__ w1,
                  const float* __restrict__ w2,
                  u16* __restrict__ wupT,
                  u16* __restrict__ w1t,
                  u16* __restrict__ w2t) {
    int idx = blockIdx.x * 256 + threadIdx.x;
    if (idx < 9 * 64 * 64) {
        int tap = idx >> 12, rem = idx & 4095;
        int oc = rem >> 6, ic = rem & 63;
        int ky = tap / 3, kx = tap % 3;
        int src = ((oc * 64 + ic) * 3 + ky) * 3 + kx;
        int dst = (tap * 64 + oc) * RS64 + ic;
        w1t[dst] = f32_to_bf16(w1[src]);
        w2t[dst] = f32_to_bf16(w2[src]);
    }
    if (idx < 4 * 64 * 128) {
        int pq = idx >> 13, rem = idx & 8191;
        int co = rem >> 7, ci = rem & 127;
        int p = pq >> 1, q = pq & 1;
        wupT[(pq * 64 + co) * RS128 + ci] =
            f32_to_bf16(w_up[((ci * 64 + co) * 2 + p) * 2 + q]);
    }
}

// ---------------------------------------------------------------------------
// Sparse merge: first N2 raster rows (= batches 0..1 of the fine grid) come
// from x2_feat.  fp32 -> bf16, two channels per dword store.
// ---------------------------------------------------------------------------
__global__ __launch_bounds__(256)
void copy_x2(const float* __restrict__ x2, u16* __restrict__ merged) {
    u32 e    = blockIdx.x * 256u + threadIdx.x;  // pair index
    u32 elem = e * 2u;
    u32 row  = elem >> 6;
    u32 c    = elem & 63u;
    u32 b    = row >> 16;
    u32 w    = row & 65535u;
    u32 y    = w >> 8, x = w & 255u;
    u32 lo = f32_to_bf16(x2[elem]);
    u32 hi = f32_to_bf16(x2[elem + 1]);
    size_t dst = (((size_t)b * HP + (y + 1)) * WP + (x + 1)) * CH + c;
    *(u32*)&merged[dst] = lo | (hi << 16);
}

// ---------------------------------------------------------------------------
// ConvTranspose2d k=2 s=2 for batches 2..3 as per-(p,q) GEMM on WMMA:
//   [16 coarse pixels x K=128] x [128 x 64]  (4 K-chunks, 4 N-tiles)
// Weights async-staged to LDS; x1 tile converted fp32->bf16 into LDS.
// ---------------------------------------------------------------------------
__global__ __launch_bounds__(256)
void up_wmma(const float* __restrict__ x1,      // [4][128][128][128] NCHW f32
             const u16*   __restrict__ wup,     // [4][64][RS128] bf16
             const float* __restrict__ bup,     // [64]
             u16*         __restrict__ merged)  // padded NHWC bf16
{
    __shared__ __align__(16) u16 s_x[32 * RS128];       //  8,704 B
    __shared__ __align__(16) u16 s_w[4 * 64 * RS128];   // 69,632 B

    int blk = blockIdx.x;
    int jt = blk & 3;   blk >>= 2;                // 4 j-tiles of 32
    int i  = blk & 127; blk >>= 7;                // coarse row
    int b  = 2 + blk;                             // batch 2..3
    const int j0 = jt * 32;

    const int tid = threadIdx.x, lane = tid & 31, wave = tid >> 5;

    // async-stage the full (p,q) weight table: 4352 x 16B contiguous
    for (int t = tid; t < 4 * 64 * RS128 / 8; t += 256)
        async_copy16(lds_off(&s_w[t * 8]), wup + t * 8);

    // stage x1 tile: [32 pixels][128 ch] with fp32->bf16 conversion
    {
        int jj = tid & 31, c0 = tid >> 5;
        for (int c = c0; c < CIN; c += 8) {
            float v = x1[(((size_t)b * CIN + c) * HH + i) * WW + (j0 + jj)];
            s_x[jj * RS128 + c] = f32_to_bf16(v);
        }
    }
    wait_async0();
    __syncthreads();

    const int pq = wave >> 1;            // 0..3 -> (p,q)
    const int mh = wave & 1;             // which 16-pixel half
    const int pixBase = mh * 16;
    const int hi = lane >> 4, ln = lane & 15;

    v8f acc[4];
    v8f z = {};
    for (int nt = 0; nt < 4; ++nt) acc[nt] = z;

    for (int kc = 0; kc < 4; ++kc) {
        FragBF A;
        const u16* ap = &s_x[(pixBase + ln) * RS128 + kc * 32 + hi * 8];
        A.q[0] = *(const u32x4*)(ap);
        A.q[1] = *(const u32x4*)(ap + 16);
        for (int nt = 0; nt < 4; ++nt) {
            FragBF Bf;
            const u16* bp = &s_w[(pq * 64 + nt * 16 + ln) * RS128 + kc * 32 + hi * 8];
            Bf.q[0] = *(const u32x4*)(bp);
            Bf.q[1] = *(const u32x4*)(bp + 16);
            acc[nt] = __builtin_amdgcn_wmma_f32_16x16x32_bf16(
                false, A.v, false, Bf.v, (short)0, acc[nt], false, false);
        }
    }

    const int p = pq >> 1, q = pq & 1;
    const int y = 2 * i + p;
    for (int nt = 0; nt < 4; ++nt) {
        const int oc = nt * 16 + ln;
        const float bv = bup[oc];
        for (int r = 0; r < 8; ++r) {
            int m = r + hi * 8;                 // D: M = vgpr + 8*(lane>=16)
            int j = j0 + pixBase + m;
            int x = 2 * j + q;
            size_t dst = (((size_t)b * HP + (y + 1)) * WP + (x + 1)) * CH + oc;
            merged[dst] = f32_to_bf16(acc[nt][r] + bv);
        }
    }
}

// ---------------------------------------------------------------------------
// 3x3 SAME conv 64->64 + bias + PReLU as implicit GEMM on WMMA.
// Block: 8x16 output pixels, all 64 oc.  Wave w -> output row y0+w.
// Input tile (10x18, padded rows) and the full 9-tap weight table are
// async-staged to LDS; both A and B fragments come from ds_load_b128.
// 72 v_wmma per wave.  mode 0: bf16 padded NHWC; mode 1: f32 NCHW.
// ---------------------------------------------------------------------------
__global__ __launch_bounds__(256)
void conv3x3_wmma(const u16*   __restrict__ in,      // padded NHWC bf16
                  const u16*   __restrict__ wt,      // [9][64][RS64] bf16
                  const float* __restrict__ bias,    // [64]
                  const float* __restrict__ alpha,   // [1] PReLU slope
                  u16*         __restrict__ out_bf,  // mode 0
                  float*       __restrict__ out_f32, // mode 1
                  int mode)
{
    __shared__ __align__(16) u16 s_in[10 * 18 * RS64];   // 25,920 B
    __shared__ __align__(16) u16 s_w[9 * 64 * RS64];     // 82,944 B

    int blk = blockIdx.x;
    int bx = blk & 15; blk >>= 4;     // 16 x-tiles of 16
    int by = blk & 31; blk >>= 5;     // 32 y-tiles of 8
    int b  = blk;
    const int x0 = bx * 16, y0 = by * 8;

    const int tid = threadIdx.x, lane = tid & 31, wave = tid >> 5;

    // async-stage input tile: 180 pixels x 128B global -> 144B-strided LDS rows
    for (int t = tid; t < 180 * 8; t += 256) {
        int p = t >> 3, d = t & 7;
        int r = p / 18, c = p % 18;
        const u16* g = in + (((size_t)b * HP + (y0 + r)) * WP + (x0 + c)) * CH + d * 8;
        async_copy16(lds_off(&s_in[p * RS64 + d * 8]), g);
    }
    // async-stage the 9-tap weight table: 5184 x 16B contiguous
    for (int t = tid; t < 9 * 64 * RS64 / 8; t += 256)
        async_copy16(lds_off(&s_w[t * 8]), wt + t * 8);
    wait_async0();
    __syncthreads();

    const int hi = lane >> 4, ln = lane & 15;
    const int oy = y0 + wave;

    v8f acc[4];
    v8f z = {};
    for (int nt = 0; nt < 4; ++nt) acc[nt] = z;

    for (int ky = 0; ky < 3; ++ky) {
        for (int kx = 0; kx < 3; ++kx) {
            const int tap = ky * 3 + kx;
            for (int kc = 0; kc < 2; ++kc) {
                FragBF A;   // 16 pixels x 32 ic
                const u16* ap = &s_in[((wave + ky) * 18 + (ln + kx)) * RS64 + kc * 32 + hi * 8];
                A.q[0] = *(const u32x4*)(ap);
                A.q[1] = *(const u32x4*)(ap + 16);
                for (int nt = 0; nt < 4; ++nt) {
                    FragBF Bf;  // 32 ic x 16 oc
                    const u16* bp = &s_w[(tap * 64 + nt * 16 + ln) * RS64 + kc * 32 + hi * 8];
                    Bf.q[0] = *(const u32x4*)(bp);
                    Bf.q[1] = *(const u32x4*)(bp + 16);
                    acc[nt] = __builtin_amdgcn_wmma_f32_16x16x32_bf16(
                        false, A.v, false, Bf.v, (short)0, acc[nt], false, false);
                }
            }
        }
    }

    const float a = alpha[0];
    for (int nt = 0; nt < 4; ++nt) {
        const int oc = nt * 16 + ln;
        const float bv = bias[oc];
        if (mode == 0) {
            for (int r = 0; r < 8; ++r) {
                int m = r + hi * 8;
                float v = acc[nt][r] + bv;
                v = v >= 0.f ? v : a * v;
                size_t dst = (((size_t)b * HP + (oy + 1)) * WP + (x0 + m + 1)) * CH + oc;
                out_bf[dst] = f32_to_bf16(v);
            }
        } else {
            f32x4 lo, hi4;
            for (int r = 0; r < 4; ++r) {
                float v = acc[nt][r] + bv;     lo[r]  = v >= 0.f ? v : a * v;
                float w = acc[nt][r + 4] + bv; hi4[r] = w >= 0.f ? w : a * w;
            }
            // x = x0 + hi*8 + r is consecutive over the vgpr index r
            float* dst = &out_f32[(((size_t)b * COUT + oc) * H2 + oy) * W2 + x0 + hi * 8];
            *(f32x4*)(dst)     = lo;
            *(f32x4*)(dst + 4) = hi4;
        }
    }
}

// ---------------------------------------------------------------------------
// Launch
// ---------------------------------------------------------------------------
extern "C" void kernel_launch(void* const* d_in, const int* in_sizes, int n_in,
                              void* d_out, int out_size, void* d_ws, size_t ws_size,
                              hipStream_t stream) {
    (void)in_sizes; (void)n_in; (void)out_size; (void)ws_size;
    const float* x1   = (const float*)d_in[0];
    const float* x2   = (const float*)d_in[1];
    const float* w_up = (const float*)d_in[2];
    const float* b_up = (const float*)d_in[3];
    const float* w1   = (const float*)d_in[4];
    const float* b1   = (const float*)d_in[5];
    const float* a1   = (const float*)d_in[6];
    const float* w2   = (const float*)d_in[7];
    const float* b2   = (const float*)d_in[8];
    const float* a2   = (const float*)d_in[9];
    float* out = (float*)d_out;

    char* ws = (char*)d_ws;
    size_t off = 0;
    const size_t padBytes = (size_t)BB * HP * WP * CH * sizeof(u16);  // 34,080,768
    u16* merged = (u16*)(ws + off); off += (padBytes + 255) & ~(size_t)255;
    u16* hbuf   = (u16*)(ws + off); off += (padBytes + 255) & ~(size_t)255;
    u16* w1t    = (u16*)(ws + off); off += ((size_t)9 * 64 * RS64 * 2 + 255) & ~(size_t)255;
    u16* w2t    = (u16*)(ws + off); off += ((size_t)9 * 64 * RS64 * 2 + 255) & ~(size_t)255;
    u16* wupT   = (u16*)(ws + off); off += ((size_t)4 * 64 * RS128 * 2 + 255) & ~(size_t)255;

    // 1. zero halos of both padded buffers (SAME padding)
    zero_halo<<<514, 256, 0, stream>>>(merged, hbuf);
    // 2. weight repack fp32 -> bf16 (padded rows)
    prep_weights<<<144, 256, 0, stream>>>(w_up, w1, w2, wupT, w1t, w2t);
    // 3. batches 0..1 of merged <- x2_feat
    copy_x2<<<16384, 256, 0, stream>>>(x2, merged);
    // 4. batches 2..3 of merged <- ConvTranspose(x1) + b_up  (WMMA)
    up_wmma<<<2 * 128 * 4, 256, 0, stream>>>(x1, wupT, b_up, merged);
    // 5. conv1 + PReLU -> hbuf (bf16 padded)
    conv3x3_wmma<<<BB * 32 * 16, 256, 0, stream>>>(merged, w1t, b1, a1, hbuf, nullptr, 0);
    // 6. conv2 + PReLU -> out (f32 NCHW)
    conv3x3_wmma<<<BB * 32 * 16, 256, 0, stream>>>(hbuf, w2t, b2, a2, nullptr, out, 1);
}
